// SEMModel_51479478010047
// MI455X (gfx1250) — compile-verified, matrix operand order checked by reading, not run on hardware
//
#include <hip/hip_runtime.h>
#include <hip/hip_bf16.h>

typedef __attribute__((ext_vector_type(16))) _Float16 v16h;
typedef __attribute__((ext_vector_type(8)))  _Float16 v8h;
typedef __attribute__((ext_vector_type(8)))  float    v8f;

#define K_B   64
#define K_T   1024
#define K_F   128
#define K_H   512
#define K_4H  2048
#define K1    (K_F + K_H)     // 640,  layer-1 K
#define K2    (2 * K_H)       // 1024, layer-2 K
#define NT_Z  (K_4H / 16)     // 128 column tiles of z
#define NGRP  8               // column groups per batch tile
#define WGSZ  256             // 8 waves (wave32)

static __device__ __forceinline__ v8f wmma_f16(v16h a, v16h b, v8f c) {
  return __builtin_amdgcn_wmma_f32_16x16x32_f16(false, a, false, b,
                                                (short)0, c, false, false);
}

static __device__ __forceinline__ float sigf(float x) {
  return 1.0f / (1.0f + __expf(-x));
}

// Build a 16x32 fp16 A-fragment for this lane from LDS (row-major, stride in halves).
// ISA 16-bit A layout: row = lane%16; element e -> K = (e/8)*16 + (lane/16)*8 + e%8.
static __device__ __forceinline__ v16h load_a_lds(const _Float16* row_base, int kbase,
                                                  int lane_grp8) {
  const v8h lo = *(const v8h*)(row_base + kbase + lane_grp8);
  const v8h hi = *(const v8h*)(row_base + kbase + lane_grp8 + 16);
  return __builtin_shufflevector(lo, hi, 0, 1, 2, 3, 4, 5, 6, 7,
                                 8, 9, 10, 11, 12, 13, 14, 15);
}

// ---------------------------------------------------------------------------
// Init: zero double-buffered h state, sync counters, loss accumulator.
// ---------------------------------------------------------------------------
__global__ void init_state(_Float16* hb1, _Float16* hb2, unsigned* cnts, float* loss) {
  const int i = blockIdx.x * blockDim.x + threadIdx.x;
  const int n = 2 * K_B * K_H;
  if (i < n) { hb1[i] = (_Float16)0.0f; hb2[i] = (_Float16)0.0f; }
  if (i < 4) cnts[i] = 0u;
  if (i == 0) *loss = 0.0f;
}

// ---------------------------------------------------------------------------
// Pack fp32 weight W[K][N] (row-major) into per-lane WMMA-B fragment order:
// dst[(((kt*NT + nt)*32 + lane)*16 + e] = W[kt*32 + (e/8)*16 + (lane/16)*8 + e%8]
//                                          [nt*16 + lane%16]
// so each lane reads its 16 halves as one contiguous 32B load.
// ---------------------------------------------------------------------------
__global__ void pack_b(const float* __restrict__ src, _Float16* __restrict__ dst,
                       int K, int N) {
  const int idx = blockIdx.x * blockDim.x + threadIdx.x;
  const int total = (K >> 5) * (N >> 4) * 512;
  if (idx >= total) return;
  const int e    = idx & 15;
  const int lane = (idx >> 4) & 31;
  const int rest = idx >> 9;
  const int NT   = N >> 4;
  const int nt   = rest % NT;
  const int kt   = rest / NT;
  const int kk   = kt * 32 + ((e >> 3) << 4) + ((lane >> 4) << 3) + (e & 7);
  const int nn   = nt * 16 + (lane & 15);
  dst[idx] = (_Float16)src[kk * N + nn];
}

// ---------------------------------------------------------------------------
// Persistent 2-layer LSTM. Grid = 4 batch tiles x 8 column groups = 32 blocks.
// Each block: 16 batch rows, 64 h-columns (i.e. 16 z-tiles across 4 gate bands).
// Cross-group sync per step: arrive counter in L2 + s_sleep spin, h double-buffered.
// Weight fragments are loop-invariant: the compiler keeps them register-resident
// across the whole T loop (VGPR-MSB file), so the inner loop is ds_load + wmma.
// ---------------------------------------------------------------------------
__global__ __launch_bounds__(WGSZ) void lstm_persistent(
    const float* __restrict__ x, const float* __restrict__ b1,
    const float* __restrict__ b2,
    const _Float16* __restrict__ pW1, const _Float16* __restrict__ pW2,
    _Float16* __restrict__ hbuf1, _Float16* __restrict__ hbuf2,
    _Float16* __restrict__ h1seq, _Float16* __restrict__ h2seq,
    unsigned* __restrict__ counters) {
  __shared__ alignas(16) _Float16 a_lds[16 * K2];   // activations (max stride 1024)
  __shared__ float z_lds[4 * 16 * 64];              // gate pre-activations, f32
  __shared__ float c_lds[16 * 64];                  // cell state slice

  const int tid  = threadIdx.x;
  const int wave = tid >> 5;
  const int lane = tid & 31;
  const int bt   = blockIdx.x >> 3;   // batch tile 0..3
  const int g    = blockIdx.x & 7;    // column group 0..7
  unsigned* cnt  = counters + bt;

  const int arow = lane & 15;               // A-fragment row
  const int agrp = (lane >> 4) << 3;        // lane half-group K offset
  const int mrow = (lane >> 4) << 3;        // C-fragment M base
  const int ncol = lane & 15;               // C-fragment N

  // z-tile assignment: wave owns tiles j0,j1 of this group's 16 tiles.
  const int j0 = wave * 2, j1 = wave * 2 + 1;
  const int tn0 = (j0 >> 2) * 32 + g * 4 + (j0 & 3);
  const int tn1 = (j1 >> 2) * 32 + g * 4 + (j1 & 3);

  for (int m = tid; m < 16 * 64; m += WGSZ) c_lds[m] = 0.0f;

  // ------------------------- LAYER 1 (K = 640) ----------------------------
  for (int s = 0; s < K_T; ++s) {
    // Stage A = [x_t (fp32->fp16), h_prev (fp16)] into LDS.
    for (int m = tid; m < 16 * K_F; m += WGSZ) {
      const int r = m >> 7, f = m & 127;
      a_lds[r * K1 + f] = (_Float16)x[((size_t)(bt * 16 + r) * K_T + s) * K_F + f];
    }
    const _Float16* hsrc = hbuf1 + (size_t)(s & 1) * K_B * K_H;
    for (int m = tid; m < 16 * (K_H / 8); m += WGSZ) {
      const int r = m >> 6, blk = m & 63;
      *(v8h*)&a_lds[r * K1 + K_F + blk * 8] =
          *(const v8h*)&hsrc[(bt * 16 + r) * K_H + blk * 8];
    }
    __syncthreads();

    v8f acc0 = {}, acc1 = {};
    const _Float16* arb = &a_lds[arow * K1];
    for (int k = 0; k < K1 / 32; ++k) {
      const v16h a  = load_a_lds(arb, k * 32, agrp);
      const v16h w0 = *(const v16h*)&pW1[(((size_t)k * NT_Z + tn0) * 32 + lane) * 16];
      const v16h w1 = *(const v16h*)&pW1[(((size_t)k * NT_Z + tn1) * 32 + lane) * 16];
      acc0 = wmma_f16(a, w0, acc0);
      acc1 = wmma_f16(a, w1, acc1);
    }
    // Stage z to LDS (C layout: M = r + 8*(lane/16), N = lane%16).
    #pragma unroll
    for (int r = 0; r < 8; ++r) {
      z_lds[(j0 >> 2) * 1024 + (r + mrow) * 64 + (j0 & 3) * 16 + ncol] = acc0[r];
      z_lds[(j1 >> 2) * 1024 + (r + mrow) * 64 + (j1 & 3) * 16 + ncol] = acc1[r];
    }
    __syncthreads();

    // Gate nonlinearity for this group's 16x64 h slice.
    for (int m = tid; m < 16 * 64; m += WGSZ) {
      const int r = m >> 6, c = m & 63;
      const int colg = g * 64 + c;
      const float iv = z_lds[0 * 1024 + r * 64 + c] + b1[colg];
      const float jv = z_lds[1 * 1024 + r * 64 + c] + b1[K_H + colg];
      const float fv = z_lds[2 * 1024 + r * 64 + c] + b1[2 * K_H + colg];
      const float ov = z_lds[3 * 1024 + r * 64 + c] + b1[3 * K_H + colg];
      const float cn = sigf(fv + 1.0f) * c_lds[m] + sigf(iv) * tanhf(jv);
      const float hv = sigf(ov) * tanhf(cn);
      c_lds[m] = cn;
      const _Float16 hh = (_Float16)hv;
      hbuf1[(size_t)((s + 1) & 1) * K_B * K_H + (bt * 16 + r) * K_H + colg] = hh;
      h1seq[((size_t)(bt * 16 + r) * K_T + s) * K_H + colg] = hh;
    }
    // Cross-group arrive/wait (per batch tile).
    __threadfence();
    __syncthreads();
    if (tid == 0) {
      __hip_atomic_fetch_add(cnt, 1u, __ATOMIC_RELEASE, __HIP_MEMORY_SCOPE_AGENT);
      const unsigned target = (unsigned)(NGRP * (s + 1));
      while (__hip_atomic_load(cnt, __ATOMIC_ACQUIRE, __HIP_MEMORY_SCOPE_AGENT) < target)
        __builtin_amdgcn_s_sleep(1);
    }
    __syncthreads();
    __threadfence();
  }

  for (int m = tid; m < 16 * 64; m += WGSZ) c_lds[m] = 0.0f;

  // ------------------------- LAYER 2 (K = 1024) ---------------------------
  for (int s = 0; s < K_T; ++s) {
    for (int m = tid; m < 16 * (K_H / 8); m += WGSZ) {
      const int r = m >> 6, blk = m & 63;
      *(v8h*)&a_lds[r * K2 + blk * 8] =
          *(const v8h*)&h1seq[((size_t)(bt * 16 + r) * K_T + s) * K_H + blk * 8];
    }
    const _Float16* hsrc = hbuf2 + (size_t)(s & 1) * K_B * K_H;
    for (int m = tid; m < 16 * (K_H / 8); m += WGSZ) {
      const int r = m >> 6, blk = m & 63;
      *(v8h*)&a_lds[r * K2 + K_H + blk * 8] =
          *(const v8h*)&hsrc[(bt * 16 + r) * K_H + blk * 8];
    }
    __syncthreads();

    v8f acc0 = {}, acc1 = {};
    const _Float16* arb = &a_lds[arow * K2];
    for (int k = 0; k < K2 / 32; ++k) {
      const v16h a  = load_a_lds(arb, k * 32, agrp);
      const v16h w0 = *(const v16h*)&pW2[(((size_t)k * NT_Z + tn0) * 32 + lane) * 16];
      const v16h w1 = *(const v16h*)&pW2[(((size_t)k * NT_Z + tn1) * 32 + lane) * 16];
      acc0 = wmma_f16(a, w0, acc0);
      acc1 = wmma_f16(a, w1, acc1);
    }
    #pragma unroll
    for (int r = 0; r < 8; ++r) {
      z_lds[(j0 >> 2) * 1024 + (r + mrow) * 64 + (j0 & 3) * 16 + ncol] = acc0[r];
      z_lds[(j1 >> 2) * 1024 + (r + mrow) * 64 + (j1 & 3) * 16 + ncol] = acc1[r];
    }
    __syncthreads();

    for (int m = tid; m < 16 * 64; m += WGSZ) {
      const int r = m >> 6, c = m & 63;
      const int colg = g * 64 + c;
      const float iv = z_lds[0 * 1024 + r * 64 + c] + b2[colg];
      const float jv = z_lds[1 * 1024 + r * 64 + c] + b2[K_H + colg];
      const float fv = z_lds[2 * 1024 + r * 64 + c] + b2[2 * K_H + colg];
      const float ov = z_lds[3 * 1024 + r * 64 + c] + b2[3 * K_H + colg];
      const float cn = sigf(fv + 1.0f) * c_lds[m] + sigf(iv) * tanhf(jv);
      const float hv = sigf(ov) * tanhf(cn);
      c_lds[m] = cn;
      const _Float16 hh = (_Float16)hv;
      hbuf2[(size_t)((s + 1) & 1) * K_B * K_H + (bt * 16 + r) * K_H + colg] = hh;
      h2seq[((size_t)(bt * 16 + r) * K_T + s) * K_H + colg] = hh;
    }
    __threadfence();
    __syncthreads();
    if (tid == 0) {
      __hip_atomic_fetch_add(cnt, 1u, __ATOMIC_RELEASE, __HIP_MEMORY_SCOPE_AGENT);
      const unsigned target = (unsigned)(NGRP * (K_T + s + 1));
      while (__hip_atomic_load(cnt, __ATOMIC_ACQUIRE, __HIP_MEMORY_SCOPE_AGENT) < target)
        __builtin_amdgcn_s_sleep(1);
    }
    __syncthreads();
    __threadfence();
  }
}

// ---------------------------------------------------------------------------
// Head: logits = h2 @ fn_w + fn_b ; accumulate sum((y - logits)^2).
// Grid = (B*T)/16 row tiles; wave w handles column tile w (F/16 = 8 tiles).
// ---------------------------------------------------------------------------
__global__ __launch_bounds__(WGSZ) void proj_loss(
    const _Float16* __restrict__ h2seq, const _Float16* __restrict__ pWf,
    const float* __restrict__ fn_b, const float* __restrict__ y,
    float* __restrict__ lossAccum) {
  __shared__ float red[WGSZ];
  const int tid  = threadIdx.x;
  const int wave = tid >> 5;          // == column tile (F/16 = 8)
  const int lane = tid & 31;
  const int rt   = blockIdx.x;        // row tile over flattened (b,t)
  const int NTf  = K_F / 16;

  const int arow = rt * 16 + (lane & 15);
  const int agrp = (lane >> 4) << 3;

  v8f acc = {};
  for (int k = 0; k < K_H / 32; ++k) {
    const _Float16* ab = &h2seq[(size_t)arow * K_H + k * 32 + agrp];
    const v8h lo = *(const v8h*)ab;
    const v8h hi = *(const v8h*)(ab + 16);
    const v16h a = __builtin_shufflevector(lo, hi, 0, 1, 2, 3, 4, 5, 6, 7,
                                           8, 9, 10, 11, 12, 13, 14, 15);
    const v16h w = *(const v16h*)&pWf[(((size_t)k * NTf + wave) * 32 + lane) * 16];
    acc = wmma_f16(a, w, acc);
  }

  const int col   = wave * 16 + (lane & 15);
  const int mbase = (lane >> 4) << 3;
  const float bcol = fn_b[col];
  float part = 0.0f;
  #pragma unroll
  for (int r = 0; r < 8; ++r) {
    const int gr = rt * 16 + mbase + r;          // = b*T + t
    const float logit = acc[r] + bcol;
    const float d = y[(size_t)gr * K_F + col] - logit;
    part += d * d;
  }

  red[tid] = part;
  __syncthreads();
  for (int s2 = WGSZ / 2; s2 > 0; s2 >>= 1) {
    if (tid < s2) red[tid] += red[tid + s2];
    __syncthreads();
  }
  if (tid == 0) atomicAdd(lossAccum, red[0]);
}

__global__ void finalize(const float* __restrict__ lossAccum, float* __restrict__ out) {
  out[0] = lossAccum[0] * (1.0f / ((float)K_B * (float)K_T * (float)K_F));
}

// ---------------------------------------------------------------------------
extern "C" void kernel_launch(void* const* d_in, const int* in_sizes, int n_in,
                              void* d_out, int out_size, void* d_ws, size_t ws_size,
                              hipStream_t stream) {
  const float* x    = (const float*)d_in[0];
  const float* y    = (const float*)d_in[1];
  const float* W1   = (const float*)d_in[2];
  const float* b1   = (const float*)d_in[3];
  const float* W2   = (const float*)d_in[4];
  const float* b2   = (const float*)d_in[5];
  const float* fn_w = (const float*)d_in[6];
  const float* fn_b = (const float*)d_in[7];
  float* out = (float*)d_out;

  // Workspace carve-up (all regions 256B aligned).
  char* base = (char*)d_ws;
  size_t off = 0;
  auto take = [&](size_t bytes) {
    char* p = base + off;
    off += (bytes + 255) & ~(size_t)255;
    return p;
  };
  _Float16* pW1   = (_Float16*)take((size_t)K1 * K_4H * 2);       // 2.5 MB
  _Float16* pW2   = (_Float16*)take((size_t)K2 * K_4H * 2);       // 4.0 MB
  _Float16* pWf   = (_Float16*)take((size_t)K_H * K_F * 2);       // 128 KB
  _Float16* hbuf1 = (_Float16*)take((size_t)2 * K_B * K_H * 2);   // 128 KB
  _Float16* hbuf2 = (_Float16*)take((size_t)2 * K_B * K_H * 2);   // 128 KB
  _Float16* h1seq = (_Float16*)take((size_t)K_B * K_T * K_H * 2); // 64 MB
  _Float16* h2seq = (_Float16*)take((size_t)K_B * K_T * K_H * 2); // 64 MB
  unsigned* cnts  = (unsigned*)take(4 * sizeof(unsigned));
  float* lossAcc  = (float*)take(sizeof(float));

  // 1) Reset recurrent state / counters / loss (required every call).
  {
    const int n = 2 * K_B * K_H;
    init_state<<<(n + 255) / 256, 256, 0, stream>>>(hbuf1, hbuf2, cnts, lossAcc);
  }
  // 2) Pack weights to fp16 WMMA-B fragment order.
  {
    const int t1 = (K1 / 32) * (K_4H / 16) * 512;
    pack_b<<<(t1 + 255) / 256, 256, 0, stream>>>(W1, pW1, K1, K_4H);
    const int t2 = (K2 / 32) * (K_4H / 16) * 512;
    pack_b<<<(t2 + 255) / 256, 256, 0, stream>>>(W2, pW2, K2, K_4H);
    const int tf = (K_H / 32) * (K_F / 16) * 512;
    pack_b<<<(tf + 255) / 256, 256, 0, stream>>>(fn_w, pWf, K_H, K_F);
  }
  // 3) Persistent 2-layer LSTM: 4 batch tiles x 8 column groups.
  lstm_persistent<<<4 * NGRP, WGSZ, 0, stream>>>(x, b1, b2, pW1, pW2, hbuf1, hbuf2,
                                                 h1seq, h2seq, cnts);
  // 4) Head GEMM + squared-error accumulation.
  proj_loss<<<(K_B * K_T) / 16, WGSZ, 0, stream>>>(h2seq, pWf, fn_b, y, lossAcc);
  // 5) Scalar mean.
  finalize<<<1, 1, 0, stream>>>(lossAcc, out);
}
